// Attention_72464688218781
// MI455X (gfx1250) — compile-verified, matrix operand order checked by reading, not run on hardware
//
#include <hip/hip_runtime.h>
#include <math.h>

#define S_LEN   2048
#define BATCH   64
#define DHID    512            // N of main GEMM
#define FTOT    1536
#define KENC    1024           // K of main GEMM (enc part only; hidden part hoisted)
#define MTOT    (S_LEN * BATCH)
#define MBLK    128            // rows per workgroup
#define NWAVE_N 2
#define NPW     (DHID / NWAVE_N)   // 256 N per wave
#define NT      (NPW / 16)         // 16 WMMA n-tiles per wave
#define PITCH   20                 // uints per 32-bf16 LDS row (64B data + 16B pad)
#define TEMP_INV (1.0f / 3.0f)

typedef __attribute__((ext_vector_type(16))) __bf16 v16bf;
typedef __attribute__((ext_vector_type(8)))  float  v8f;
typedef __attribute__((ext_vector_type(4)))  float  v4f;
typedef __attribute__((ext_vector_type(4)))  int    v4i;

union FragBF {
    uint4 u4[2];
    v16bf v;
};

#if __has_builtin(__builtin_amdgcn_global_load_async_to_lds_b128) && \
    __has_builtin(__builtin_amdgcn_s_wait_asynccnt)
#define USE_ASYNC_LDS 1
#else
#define USE_ASYNC_LDS 0
#endif

#if USE_ASYNC_LDS
typedef __attribute__((address_space(1))) v4i g1_v4i;   // global addrspace
typedef __attribute__((address_space(3))) v4i l3_v4i;   // LDS addrspace
#endif

__device__ __forceinline__ void copy_b128_to_lds(const uint4* gsrc, unsigned int* ldst) {
#if USE_ASYNC_LDS
    __builtin_amdgcn_global_load_async_to_lds_b128(
        (g1_v4i*)gsrc, (l3_v4i*)ldst, 0, 0);
#else
    *(uint4*)ldst = *gsrc;
#endif
}

__device__ __forceinline__ void wait_async_copies() {
#if USE_ASYNC_LDS
    __builtin_amdgcn_s_wait_asynccnt(0);
#endif
}

__device__ __forceinline__ unsigned int f32_to_bf16_bits(float f) {
    unsigned int x = __float_as_uint(f);
    unsigned int r = x + 0x7FFFu + ((x >> 16) & 1u);   // round-to-nearest-even
    return r >> 16;
}

// Saturating fast tanh: exact limits at +/-inf, hardware v_exp_f32 core.
__device__ __forceinline__ float fast_tanhf(float x) {
    float e = __expf(2.0f * x);
    return 1.0f - 2.0f / (e + 1.0f);
}

// ---------------------------------------------------------------------------
// Kernel 0a: hpart[b,d] = attn_b[d] + sum_f<512 hidden[b,f] * attn_w[d,f]
// ---------------------------------------------------------------------------
__global__ __launch_bounds__(256)
void hpart_kernel(const float* __restrict__ hidden,
                  const float* __restrict__ attn_w,
                  const float* __restrict__ attn_b,
                  float* __restrict__ hpart) {
    const int b = blockIdx.x;
    const float* hrow = hidden + (size_t)b * DHID;
    for (int d = threadIdx.x; d < DHID; d += 256) {
        const float* wrow = attn_w + (size_t)d * FTOT;
        float s = 0.0f;
        #pragma unroll 8
        for (int f = 0; f < DHID; ++f) s += hrow[f] * wrow[f];
        hpart[(size_t)b * DHID + d] = s + attn_b[d];
    }
}

// ---------------------------------------------------------------------------
// Kernel 0b: repack attn_w[:, 512:1536] -> bf16, K-step-tiled layout:
//   element (d, k) -> w2bf[ ((k>>5)*512 + d)*32 + (k&31) ]
// so every 32-wide K-step is one contiguous 32 KB block.
// ---------------------------------------------------------------------------
__global__ __launch_bounds__(256)
void w2_convert_kernel(const float* __restrict__ attn_w,
                       unsigned short* __restrict__ w2bf) {
    const int flat = (blockIdx.x * 256 + threadIdx.x) * 8;   // over 512*1024
    const int d = flat >> 10;
    const int k = flat & 1023;
    const float4* src = (const float4*)(attn_w + (size_t)d * FTOT + DHID + k);
    float4 x0 = src[0];
    float4 x1 = src[1];
    unsigned int p0 = f32_to_bf16_bits(x0.x) | (f32_to_bf16_bits(x0.y) << 16);
    unsigned int p1 = f32_to_bf16_bits(x0.z) | (f32_to_bf16_bits(x0.w) << 16);
    unsigned int p2 = f32_to_bf16_bits(x1.x) | (f32_to_bf16_bits(x1.y) << 16);
    unsigned int p3 = f32_to_bf16_bits(x1.z) | (f32_to_bf16_bits(x1.w) << 16);
    uint4 outv = make_uint4(p0, p1, p2, p3);
    const size_t dst = ((size_t)(k >> 5) * DHID + d) * 32 + (k & 31);
    *(uint4*)(w2bf + dst) = outv;
}

// ---------------------------------------------------------------------------
// Main kernel: workgroup = 128 rows (m = s*64+b) x full N=512 via bf16 WMMA.
// 8 waves tiled 4(M) x 2(N); each wave: 2 A-fragments x 16 B-tiles = 32 C
// tiles, so every B fragment feeds two WMMAs. Double-buffered LDS (80B row
// pitch to kill 4-way bank conflicts), async global->LDS B staging, fused
// tanh + v-dot epilogue -> attn[m].
// ---------------------------------------------------------------------------
__global__ __launch_bounds__(256)
void energy_attn_kernel(const float* __restrict__ enc,
                        const float* __restrict__ hpart,
                        const unsigned short* __restrict__ w2bf,
                        const float* __restrict__ v_w,
                        const float* __restrict__ v_b,
                        float* __restrict__ attn) {
    __shared__ __align__(16) unsigned int ldsB[2][DHID * PITCH];  // 2 x 40 KB
    __shared__ __align__(16) unsigned int ldsA[2][MBLK * PITCH];  // 2 x 10 KB
    __shared__ float part[MBLK];

    const int tid  = threadIdx.x;
    const int lane = tid & 31;
    const int wave = tid >> 5;
    const int wm   = wave >> 1;      // 0..3 -> rows wm*32 .. wm*32+31
    const int wn   = wave & 1;       // 0..1 -> N slice wn*256
    const int half = lane >> 4;      // 0/1
    const int lr   = lane & 15;
    const int mbase = blockIdx.x * MBLK;
    const int n0 = wn * NPW;

    if (tid < MBLK) part[tid] = 0.0f;

    v8f acc[2][NT];                  // [A-fragment set][n-tile]
    const v8f vzero = {0.f, 0.f, 0.f, 0.f, 0.f, 0.f, 0.f, 0.f};
    #pragma unroll
    for (int h = 0; h < 2; ++h)
        #pragma unroll
        for (int t = 0; t < NT; ++t) acc[h][t] = vzero;

    // A-staging duty: thread -> row tid>>1, 16 consecutive floats at col (tid&1)*16
    const int arow = tid >> 1;                       // 0..127
    const int acol = (tid & 1) * 16;
    const float* asrc = enc + (size_t)(mbase + arow) * KENC + acol;
    const int adst = arow * PITCH + (tid & 1) * 8;   // uint index inside a stage

    // B-staging duty: chunk c = tid + 256*i; row = c>>2 (0..511), quarter = c&3
    const int brow0 = tid >> 2;
    const int bq    = tid & 3;
    const int bdst0 = brow0 * PITCH + bq * 4;        // +64*PITCH per i

    const uint4* bsrcbase = (const uint4*)w2bf;

    auto stage = [&](int p, int kk) {
        // --- A tile: 128x32 f32 -> bf16 (streaming loads, keep L2 for B) ---
        v4f x0 = __builtin_nontemporal_load((const v4f*)(asrc + kk));
        v4f x1 = __builtin_nontemporal_load((const v4f*)(asrc + kk) + 1);
        v4f x2 = __builtin_nontemporal_load((const v4f*)(asrc + kk) + 2);
        v4f x3 = __builtin_nontemporal_load((const v4f*)(asrc + kk) + 3);
        uint4 pk0, pk1;
        pk0.x = f32_to_bf16_bits(x0.x) | (f32_to_bf16_bits(x0.y) << 16);
        pk0.y = f32_to_bf16_bits(x0.z) | (f32_to_bf16_bits(x0.w) << 16);
        pk0.z = f32_to_bf16_bits(x1.x) | (f32_to_bf16_bits(x1.y) << 16);
        pk0.w = f32_to_bf16_bits(x1.z) | (f32_to_bf16_bits(x1.w) << 16);
        pk1.x = f32_to_bf16_bits(x2.x) | (f32_to_bf16_bits(x2.y) << 16);
        pk1.y = f32_to_bf16_bits(x2.z) | (f32_to_bf16_bits(x2.w) << 16);
        pk1.z = f32_to_bf16_bits(x3.x) | (f32_to_bf16_bits(x3.y) << 16);
        pk1.w = f32_to_bf16_bits(x3.z) | (f32_to_bf16_bits(x3.w) << 16);
        *(uint4*)&ldsA[p][adst]     = pk0;
        *(uint4*)&ldsA[p][adst + 4] = pk1;
        // --- B strip: 512x32 bf16 = 32 KB source, async to padded LDS rows ---
        const uint4* bsrc = bsrcbase + (size_t)(kk >> 5) * 2048 + tid;
        #pragma unroll
        for (int i = 0; i < 8; ++i)
            copy_b128_to_lds(bsrc + 256 * i, &ldsB[p][bdst0 + i * 64 * PITCH]);
    };

    auto compute = [&](int p) {
        FragBF a0, a1;
        const int ar0 = (wm * 32 + lr) * PITCH;
        const int ar1 = (wm * 32 + 16 + lr) * PITCH;
        a0.u4[0] = *(const uint4*)&ldsA[p][ar0 + half * 4];
        a0.u4[1] = *(const uint4*)&ldsA[p][ar0 + 8 + half * 4];
        a1.u4[0] = *(const uint4*)&ldsA[p][ar1 + half * 4];
        a1.u4[1] = *(const uint4*)&ldsA[p][ar1 + 8 + half * 4];
        #pragma unroll
        for (int t = 0; t < NT; ++t) {
            const int n = (n0 + t * 16 + lr) * PITCH;
            FragBF bfrag;
            bfrag.u4[0] = *(const uint4*)&ldsB[p][n + half * 4];
            bfrag.u4[1] = *(const uint4*)&ldsB[p][n + 8 + half * 4];
            // one B fragment feeds two WMMAs (second has no new LDS dep)
            acc[0][t] = __builtin_amdgcn_wmma_f32_16x16x32_bf16(
                false, a0.v, false, bfrag.v, (short)0, acc[0][t], false, false);
            acc[1][t] = __builtin_amdgcn_wmma_f32_16x16x32_bf16(
                false, a1.v, false, bfrag.v, (short)0, acc[1][t], false, false);
        }
    };

    // Prologue: stage k-step 0
    stage(0, 0);
    wait_async_copies();
    __syncthreads();

    int p = 0;
    for (int kk = 0; kk < KENC; kk += 32) {
        if (kk + 32 < KENC) stage(p ^ 1, kk + 32);  // overlap fetch with WMMA
        compute(p);
        wait_async_copies();
        __syncthreads();
        p ^= 1;
    }

    // --- fused epilogue: energy = tanh(acc + hpart); contrib = v_w * energy ---
    const float vb = v_b[0];
    #pragma unroll
    for (int h = 0; h < 2; ++h) {
        #pragma unroll
        for (int r = 0; r < 8; ++r) {
            const int mloc = wm * 32 + h * 16 + half * 8 + r;  // C layout: M = half*8 + vgpr
            const int brow = mloc & (BATCH - 1);               // b = mloc % 64
            float contrib = 0.0f;
            #pragma unroll
            for (int t = 0; t < NT; ++t) {
                const int d = n0 + t * 16 + lr;                // C layout: N = lane%16
                float pre = acc[h][t][r] + hpart[(size_t)brow * DHID + d];
                contrib += v_w[d] * fast_tanhf(pre);
            }
            #pragma unroll
            for (int off = 1; off < 16; off <<= 1)
                contrib += __shfl_xor(contrib, off, 32);
            if (lr == 0) atomicAdd(&part[mloc], contrib);
        }
    }
    __syncthreads();
    if (tid < MBLK) attn[mbase + tid] = part[tid] + vb;
}

// ---------------------------------------------------------------------------
// Softmax over S with temperature (matches reference: no max-subtraction)
// attn layout: [s][b] (m = s*64 + b); out identical layout.
// ---------------------------------------------------------------------------
__global__ __launch_bounds__(256)
void softmax_kernel(const float* __restrict__ attn, float* __restrict__ out) {
    __shared__ float red[8];
    __shared__ float totsh;
    const int b = blockIdx.x;
    float e[8];
    float sum = 0.0f;
    #pragma unroll
    for (int i = 0; i < 8; ++i) {
        const int s = threadIdx.x + i * 256;
        e[i] = __expf(attn[(size_t)s * BATCH + b] * TEMP_INV);
        sum += e[i];
    }
    #pragma unroll
    for (int off = 1; off < 32; off <<= 1) sum += __shfl_xor(sum, off, 32);
    const int wv = threadIdx.x >> 5;
    const int ln = threadIdx.x & 31;
    if (ln == 0) red[wv] = sum;
    __syncthreads();
    if (threadIdx.x < 8) {
        float v = red[threadIdx.x];
        #pragma unroll
        for (int off = 1; off < 8; off <<= 1) v += __shfl_xor(v, off, 32);
        if (threadIdx.x == 0) totsh = v;
    }
    __syncthreads();
    const float inv = 1.0f / totsh;
    #pragma unroll
    for (int i = 0; i < 8; ++i) {
        const int s = threadIdx.x + i * 256;
        out[(size_t)s * BATCH + b] = e[i] * inv;
    }
}

// ---------------------------------------------------------------------------
extern "C" void kernel_launch(void* const* d_in, const int* in_sizes, int n_in,
                              void* d_out, int out_size, void* d_ws, size_t ws_size,
                              hipStream_t stream) {
    const float* hidden = (const float*)d_in[0];   // [64, 512]
    const float* enc    = (const float*)d_in[1];   // [2048, 64, 1024]
    const float* attn_w = (const float*)d_in[2];   // [512, 1536]
    const float* attn_b = (const float*)d_in[3];   // [512]
    const float* v_w    = (const float*)d_in[4];   // [1, 512]
    const float* v_b    = (const float*)d_in[5];   // [1]
    float* out = (float*)d_out;                    // [2048, 64, 1]

    char* ws = (char*)d_ws;
    unsigned short* w2bf = (unsigned short*)ws;                       // 1 MB
    float* hpart   = (float*)(ws + 1024 * 1024);                      // 128 KB
    float* attnbuf = (float*)(ws + 1024 * 1024 + BATCH * DHID * 4);   // 512 KB

    w2_convert_kernel<<<(DHID * KENC) / (8 * 256), 256, 0, stream>>>(attn_w, w2bf);
    hpart_kernel<<<BATCH, 256, 0, stream>>>(hidden, attn_w, attn_b, hpart);
    energy_attn_kernel<<<MTOT / MBLK, 256, 0, stream>>>(enc, hpart, w2bf, v_w, v_b, attnbuf);
    softmax_kernel<<<BATCH, 256, 0, stream>>>(attnbuf, out);
}
